// GNN_17669495456025
// MI455X (gfx1250) — matrix operand order repaired and checked by reading, not compiled, not run on hardware
//
#include <hip/hip_runtime.h>

#define Nn  100000
#define Ee  1600000
#define Gg  256
#define NFE 7
#define EFE 3
#define UDm 2
#define Hd  128
#define NLs 2

// f16-image sizes (in halves) per layer, in WMMA-friendly [n][k] layout
#define W1_ELEMS 4096   // [128][32]
#define W2_ELEMS 2048   // [16][128]

typedef __attribute__((ext_vector_type(16))) _Float16 v16h;
typedef __attribute__((ext_vector_type(8)))  _Float16 v8h;
typedef __attribute__((ext_vector_type(8)))  float    v8f;

__device__ __forceinline__ void atomicMaxF(float* addr, float val) {
  // int-max for non-negative, uint-min for negative; works with -inf init
  if (val >= 0.0f) atomicMax((int*)addr, __float_as_int(val));
  else             atomicMin((unsigned int*)addr, __float_as_uint(val));
}

// One wave: 16 rows, MLP  out16x16 = relu(A[16x32] * W1[32x128] + b1) * W2[128x16] + b2
__device__ __forceinline__ v8f wave_mlp16(const _Float16* sAw, const _Float16* sW1,
                                          const float* sB1, _Float16* sHw,
                                          const _Float16* sW2, const float* sB2, int lane) {
  const int m      = lane & 15;
  const int hiHalf = (lane >> 4) & 1;
  // A fragment: lanes 0-15 hold K 0-7 / 16-23, lanes 16-31 hold K 8-15 / 24-31
  const int kA0 = hiHalf ? 8 : 0;
  const int kA1 = hiHalf ? 24 : 16;
  v8h alo = *(const v8h*)(sAw + m * 32 + kA0);
  v8h ahi = *(const v8h*)(sAw + m * 32 + kA1);
  v16h a;
#pragma unroll
  for (int i = 0; i < 8; ++i) { a[i] = alo[i]; a[8 + i] = ahi[i]; }
  // B: same K split as A (lanes 0-15 K 0-7 / 16-23, lanes 16-31 K 8-15 / 24-31)

  // ---- layer 1: hidden[16x128] over 8 N-tiles, bias in C, ReLU fused on spill ----
#pragma unroll
  for (int t = 0; t < 8; ++t) {
    const int n = t * 16 + m;
    v8h blo = *(const v8h*)(sW1 + n * 32 + kA0);
    v8h bhi = *(const v8h*)(sW1 + n * 32 + kA1);
    v16h b;
#pragma unroll
    for (int i = 0; i < 8; ++i) { b[i] = blo[i]; b[8 + i] = bhi[i]; }
    float bias = sB1[n];
    v8f c;
#pragma unroll
    for (int v = 0; v < 8; ++v) c[v] = bias;
    c = __builtin_amdgcn_wmma_f32_16x16x32_f16(false, a, false, b, (short)0, c, false, false);
#pragma unroll
    for (int v = 0; v < 8; ++v) {
      int M = v + (hiHalf ? 8 : 0);
      float h = c[v] > 0.0f ? c[v] : 0.0f;
      sHw[M * 128 + t * 16 + m] = (_Float16)h;
    }
  }
  // ---- layer 2: K=128 in 4 WMMA chunks, N padded to 16 ----
  float bias2 = sB2[m];
  v8f c2;
#pragma unroll
  for (int v = 0; v < 8; ++v) c2[v] = bias2;
#pragma unroll
  for (int kc = 0; kc < 4; ++kc) {
    int klo = kc * 32 + (hiHalf ? 8 : 0);
    int khi = kc * 32 + (hiHalf ? 24 : 16);
    v8h a2lo = *(const v8h*)(sHw + m * 128 + klo);
    v8h a2hi = *(const v8h*)(sHw + m * 128 + khi);
    v16h a2;
#pragma unroll
    for (int i = 0; i < 8; ++i) { a2[i] = a2lo[i]; a2[8 + i] = a2hi[i]; }
    v8h b2lo = *(const v8h*)(sW2 + m * 128 + klo);
    v8h b2hi = *(const v8h*)(sW2 + m * 128 + khi);
    v16h b2;
#pragma unroll
    for (int i = 0; i < 8; ++i) { b2[i] = b2lo[i]; b2[8 + i] = b2hi[i]; }
    c2 = __builtin_amdgcn_wmma_f32_16x16x32_f16(false, a2, false, b2, (short)0, c2, false, false);
  }
  return c2;
}

// One-time (per launch) conversion of all MLP weights into f16 images in the exact
// LDS [n][k] layout the WMMA fragments read; blocks then copy them with b128 ops.
__global__ void prep_weights_kernel(const float* __restrict__ eW1, const float* __restrict__ eW2,
                                    const float* __restrict__ nW1, const float* __restrict__ nW2,
                                    _Float16* __restrict__ wE1, _Float16* __restrict__ wE2,
                                    _Float16* __restrict__ wN1, _Float16* __restrict__ wN2) {
  int i = blockIdx.x * blockDim.x + threadIdx.x;
  const int nE1 = NLs * W1_ELEMS;          // 8192
  const int nE2 = nE1 + NLs * W2_ELEMS;    // 12288
  const int nN1 = nE2 + NLs * W1_ELEMS;    // 20480
  const int nN2 = nN1 + NLs * W2_ELEMS;    // 24576
  if (i < nE1) {                                        // eW1 [l][17][128] -> [l][n=128][k=32]
    int l = i >> 12, r = i & 4095, n = r >> 5, k = r & 31;
    wE1[i] = (k < 17) ? (_Float16)eW1[l * 17 * Hd + k * Hd + n] : (_Float16)0.0f;
  } else if (i < nE2) {                                 // eW2 [l][128][3] -> [l][n=16][k=128]
    int j = i - nE1, l = j >> 11, r = j & 2047, n = r >> 7, k = r & 127;
    wE2[j] = (n < 3) ? (_Float16)eW2[l * Hd * EFE + k * EFE + n] : (_Float16)0.0f;
  } else if (i < nN1) {                                 // nW1 [l][18][128]
    int j = i - nE2, l = j >> 12, r = j & 4095, n = r >> 5, k = r & 31;
    wN1[j] = (k < 18) ? (_Float16)nW1[l * 18 * Hd + k * Hd + n] : (_Float16)0.0f;
  } else if (i < nN2) {                                 // nW2 [l][128][7]
    int j = i - nN1, l = j >> 11, r = j & 2047, n = r >> 7, k = r & 127;
    wN2[j] = (n < 7) ? (_Float16)nW2[l * Hd * NFE + k * NFE + n] : (_Float16)0.0f;
  }
}

__global__ void zero_node_kernel(float* nsum, float* nmax, float* ncnt) {
  int i = blockIdx.x * blockDim.x + threadIdx.x;
  if (i < Nn) {
#pragma unroll
    for (int f = 0; f < EFE; ++f) { nsum[i * EFE + f] = 0.0f; nmax[i * EFE + f] = -__builtin_inff(); }
    ncnt[i] = 0.0f;
  }
}

__global__ void zero_graph_kernel(float* gsum, float* gmax, float* gcnt) {
  int g = threadIdx.x;
  if (g < Gg) {
#pragma unroll
    for (int f = 0; f < NFE; ++f) { gsum[g * NFE + f] = 0.0f; gmax[g * NFE + f] = -__builtin_inff(); }
    gcnt[g] = 0.0f;
  }
}

// EdgeModel: e_out = MLP(concat(x[row], x[col], e_in)) + scatter(sum/max/cnt by col)
// 256 threads = 8 waves, 128 edges per block
__global__ __launch_bounds__(256)
void edge_kernel(const float* __restrict__ x_in, const float* __restrict__ e_in,
                 const _Float16* __restrict__ W1h, const float* __restrict__ b1,
                 const _Float16* __restrict__ W2h, const float* __restrict__ b2,
                 const int* __restrict__ row, const int* __restrict__ col,
                 float* __restrict__ e_out,
                 float* __restrict__ nsum, float* __restrict__ nmax, float* __restrict__ ncnt) {
  __shared__ alignas(64) _Float16 sW1[W1_ELEMS];
  __shared__ alignas(64) _Float16 sW2[W2_ELEMS];
  __shared__ alignas(64) _Float16 sA[128 * 32];
  __shared__ alignas(64) _Float16 sH[8 * 16 * 128];
  __shared__ float sB1[128];
  __shared__ float sB2[16];
  __shared__ int   sRow[128], sCol[128];

  const int tid   = threadIdx.x;
  const int ebase = blockIdx.x * 128;

  __builtin_prefetch(W1h, 0, 1);             // global_prefetch_b8: pull weight image
  __builtin_prefetch(W2h, 0, 1);

  // b128 copy of pre-layouted f16 weights: 512+256 uint4 over 256 threads
  {
    const uint4* g1 = (const uint4*)W1h; uint4* l1 = (uint4*)sW1;
#pragma unroll
    for (int i = 0; i < 2; ++i) l1[tid + i * 256] = g1[tid + i * 256];
    const uint4* g2 = (const uint4*)W2h; uint4* l2 = (uint4*)sW2;
    l2[tid] = g2[tid];
  }
  if (tid < 128) sB1[tid] = b1[tid];
  if (tid < 16)  sB2[tid] = (tid < 3) ? b2[tid] : 0.0f;
  if (tid < 128) {
    int e = ebase + tid;
    sRow[tid] = row[e];
    int c = col[e];
    sCol[tid] = c;
    atomicAdd(&ncnt[c], 1.0f);
  }
  __syncthreads();

  for (int idx = tid; idx < 128 * 32; idx += 256) {   // gather inputs -> f16 LDS (pad K)
    int el = idx >> 5, k = idx & 31;
    int e = ebase + el;
    float v;
    if (k < 7)       v = x_in[sRow[el] * NFE + k];
    else if (k < 14) v = x_in[sCol[el] * NFE + (k - 7)];
    else if (k < 17) v = e_in[e * EFE + (k - 14)];
    else             v = 0.0f;
    sA[el * 32 + k] = (_Float16)v;
  }
  __syncthreads();

  const int w = tid >> 5, lane = tid & 31;
  v8f c2 = wave_mlp16(sA + w * 16 * 32, sW1, sB1, sH + w * 16 * 128, sW2, sB2, lane);

  const int n2 = lane & 15;
  if (n2 < 3) {
    const int hiHalf = (lane >> 4) & 1;
#pragma unroll
    for (int v = 0; v < 8; ++v) {
      int M = v + (hiHalf ? 8 : 0);
      int el = w * 16 + M;
      int e = ebase + el;
      float val = c2[v];
      e_out[e * EFE + n2] = val;
      int cn = sCol[el];
      atomicAdd(&nsum[cn * EFE + n2], val);
      atomicMaxF(&nmax[cn * EFE + n2], val);
    }
  }
}

// NodeModel: x_out = MLP(concat(x, sum, max, mean, u[batch])); 128 nodes per block
__global__ __launch_bounds__(256)
void node_kernel(const float* __restrict__ x_in,
                 const float* __restrict__ nsum, const float* __restrict__ nmax,
                 const float* __restrict__ ncnt,
                 const float* __restrict__ u, const int* __restrict__ batch,
                 const _Float16* __restrict__ W1h, const float* __restrict__ b1,
                 const _Float16* __restrict__ W2h, const float* __restrict__ b2,
                 float* __restrict__ x_out) {
  __shared__ alignas(64) _Float16 sW1[W1_ELEMS];
  __shared__ alignas(64) _Float16 sW2[W2_ELEMS];
  __shared__ alignas(64) _Float16 sA[128 * 32];
  __shared__ alignas(64) _Float16 sH[8 * 16 * 128];
  __shared__ float sB1[128];
  __shared__ float sB2[16];

  const int tid   = threadIdx.x;
  const int nbase = blockIdx.x * 128;

  __builtin_prefetch(W1h, 0, 1);
  __builtin_prefetch(W2h, 0, 1);
  {
    const uint4* g1 = (const uint4*)W1h; uint4* l1 = (uint4*)sW1;
#pragma unroll
    for (int i = 0; i < 2; ++i) l1[tid + i * 256] = g1[tid + i * 256];
    const uint4* g2 = (const uint4*)W2h; uint4* l2 = (uint4*)sW2;
    l2[tid] = g2[tid];
  }
  if (tid < 128) sB1[tid] = b1[tid];
  if (tid < 16)  sB2[tid] = (tid < 7) ? b2[tid] : 0.0f;
  __syncthreads();

  for (int idx = tid; idx < 128 * 32; idx += 256) {
    int nl = idx >> 5, k = idx & 31;
    int i = nbase + nl; if (i >= Nn) i = Nn - 1;       // clamp tail (stores are guarded)
    float v;
    if (k < 7)       v = x_in[i * NFE + k];
    else if (k < 10) v = nsum[i * EFE + (k - 7)];
    else if (k < 13) { float c = ncnt[i]; float mv = nmax[i * EFE + (k - 10)]; v = (c > 0.0f) ? mv : 0.0f; }
    else if (k < 16) { float c = ncnt[i]; v = nsum[i * EFE + (k - 13)] / fmaxf(c, 1.0f); }
    else if (k < 18) v = u[batch[i] * UDm + (k - 16)];
    else             v = 0.0f;
    sA[nl * 32 + k] = (_Float16)v;
  }
  __syncthreads();

  const int w = tid >> 5, lane = tid & 31;
  v8f c2 = wave_mlp16(sA + w * 16 * 32, sW1, sB1, sH + w * 16 * 128, sW2, sB2, lane);

  const int n2 = lane & 15;
  if (n2 < 7) {
    const int hiHalf = (lane >> 4) & 1;
#pragma unroll
    for (int v = 0; v < 8; ++v) {
      int M = v + (hiHalf ? 8 : 0);
      int i = nbase + w * 16 + M;
      if (i < Nn) x_out[i * NFE + n2] = c2[v];
    }
  }
}

__global__ void pool_kernel(const float* __restrict__ xf, const int* __restrict__ batch,
                            float* __restrict__ gsum, float* __restrict__ gmax,
                            float* __restrict__ gcnt) {
  int i = blockIdx.x * blockDim.x + threadIdx.x;
  if (i < Nn) {
    int g = batch[i];
#pragma unroll
    for (int f = 0; f < NFE; ++f) {
      float v = xf[i * NFE + f];
      atomicAdd(&gsum[g * NFE + f], v);
      atomicMaxF(&gmax[g * NFE + f], v);
    }
    atomicAdd(&gcnt[g], 1.0f);
  }
}

// Head: [G,23] -> 128 -> 128 -> 128 -> 1   (tiny; scalar per-graph block)
__global__ __launch_bounds__(128)
void head_kernel(const float* __restrict__ gsum, const float* __restrict__ gmax,
                 const float* __restrict__ gcnt, const float* __restrict__ u,
                 const float* __restrict__ oW1, const float* __restrict__ ob1,
                 const float* __restrict__ oW2, const float* __restrict__ ob2,
                 const float* __restrict__ oW3, const float* __restrict__ ob3,
                 const float* __restrict__ oW4, const float* __restrict__ ob4,
                 float* __restrict__ out) {
  __shared__ float sin23[23];
  __shared__ float sh[128];
  __shared__ float sred[128];
  const int g = blockIdx.x, j = threadIdx.x;
  if (j == 0) {
    float c = gcnt[g];
    for (int f = 0; f < NFE; ++f) {
      float s = gsum[g * NFE + f];
      sin23[f]           = s;                       // sum
      sin23[NFE + f]     = s / fmaxf(c, 1.0f);      // mean
      float mv = gmax[g * NFE + f];
      sin23[2 * NFE + f] = (c > 0.0f) ? mv : 0.0f;  // max
    }
    sin23[21] = u[g * UDm];
    sin23[22] = u[g * UDm + 1];
  }
  __syncthreads();
  float h = ob1[j];
  for (int k = 0; k < 23; ++k) h += sin23[k] * oW1[k * 128 + j];
  h = fmaxf(h, 0.0f);
  sh[j] = h; __syncthreads();
  float h2 = ob2[j];
  for (int k = 0; k < 128; ++k) h2 += sh[k] * oW2[k * 128 + j];
  h2 = fmaxf(h2, 0.0f);
  __syncthreads(); sh[j] = h2; __syncthreads();
  float h3 = ob3[j];
  for (int k = 0; k < 128; ++k) h3 += sh[k] * oW3[k * 128 + j];
  h3 = fmaxf(h3, 0.0f);
  sred[j] = h3 * oW4[j];
  __syncthreads();
  for (int s = 64; s > 0; s >>= 1) { if (j < s) sred[j] += sred[j + s]; __syncthreads(); }
  if (j == 0) out[g] = sred[0] + ob4[0];
}

extern "C" void kernel_launch(void* const* d_in, const int* in_sizes, int n_in,
                              void* d_out, int out_size, void* d_ws, size_t ws_size,
                              hipStream_t stream) {
  (void)in_sizes; (void)n_in; (void)out_size; (void)ws_size;
  const float* x   = (const float*)d_in[0];
  const float* ea  = (const float*)d_in[1];
  const float* u   = (const float*)d_in[2];
  const float* eW1 = (const float*)d_in[3];
  const float* eb1 = (const float*)d_in[4];
  const float* eW2 = (const float*)d_in[5];
  const float* eb2 = (const float*)d_in[6];
  const float* nW1 = (const float*)d_in[7];
  const float* nb1 = (const float*)d_in[8];
  const float* nW2 = (const float*)d_in[9];
  const float* nb2 = (const float*)d_in[10];
  const float* oW1 = (const float*)d_in[11];
  const float* ob1 = (const float*)d_in[12];
  const float* oW2 = (const float*)d_in[13];
  const float* ob2 = (const float*)d_in[14];
  const float* oW3 = (const float*)d_in[15];
  const float* ob3 = (const float*)d_in[16];
  const float* oW4 = (const float*)d_in[17];
  const float* ob4 = (const float*)d_in[18];
  const int*   ei  = (const int*)d_in[19];
  const int* batch = (const int*)d_in[20];
  const int* row = ei;
  const int* col = ei + Ee;

  float* ws   = (float*)d_ws;
  float* xbuf = ws;  ws += Nn * NFE;          // node features (in-place across layers)
  float* ebuf = ws;  ws += (size_t)Ee * EFE;  // edge features (in-place across layers)
  float* nsum = ws;  ws += Nn * EFE;
  float* nmax = ws;  ws += Nn * EFE;
  float* ncnt = ws;  ws += Nn;
  float* gsum = ws;  ws += Gg * NFE;
  float* gmax = ws;  ws += Gg * NFE;
  float* gcnt = ws;  ws += Gg;
  // f16 weight images (prefix sums above keep this 16B-aligned)
  _Float16* wE1 = (_Float16*)ws;
  _Float16* wE2 = wE1 + NLs * W1_ELEMS;
  _Float16* wN1 = wE2 + NLs * W2_ELEMS;
  _Float16* wN2 = wN1 + NLs * W1_ELEMS;

  const int EB = Ee / 128;            // 12500 blocks, exact
  const int NB = (Nn + 127) / 128;    // 782 blocks

  hipLaunchKernelGGL(prep_weights_kernel, dim3(96), dim3(256), 0, stream,
                     eW1, eW2, nW1, nW2, wE1, wE2, wN1, wN2);

  const float* xs = x;
  const float* es = ea;
  for (int l = 0; l < NLs; ++l) {
    hipLaunchKernelGGL(zero_node_kernel, dim3((Nn + 255) / 256), dim3(256), 0, stream,
                       nsum, nmax, ncnt);
    hipLaunchKernelGGL(edge_kernel, dim3(EB), dim3(256), 0, stream,
                       xs, es, wE1 + l * W1_ELEMS, eb1 + l * Hd, wE2 + l * W2_ELEMS, eb2 + l * EFE,
                       row, col, ebuf, nsum, nmax, ncnt);
    hipLaunchKernelGGL(node_kernel, dim3(NB), dim3(256), 0, stream,
                       xs, nsum, nmax, ncnt, u, batch,
                       wN1 + l * W1_ELEMS, nb1 + l * Hd, wN2 + l * W2_ELEMS, nb2 + l * NFE, xbuf);
    xs = xbuf; es = ebuf;
  }
  hipLaunchKernelGGL(zero_graph_kernel, dim3(1), dim3(256), 0, stream, gsum, gmax, gcnt);
  hipLaunchKernelGGL(pool_kernel, dim3((Nn + 255) / 256), dim3(256), 0, stream,
                     xbuf, batch, gsum, gmax, gcnt);
  hipLaunchKernelGGL(head_kernel, dim3(Gg), dim3(128), 0, stream,
                     gsum, gmax, gcnt, u, oW1, ob1, oW2, ob2, oW3, ob3, oW4, ob4,
                     (float*)d_out);
}
